// GaborFilter_36395552866978
// MI455X (gfx1250) — matrix operand order repaired and checked by reading, not compiled, not run on hardware
//
#include <hip/hip_runtime.h>
#include <math.h>

typedef __attribute__((ext_vector_type(16))) _Float16 v16h;
typedef __attribute__((ext_vector_type(8)))  float    v8f;

#define PI_F 3.14159265358979323846f

static constexpr long long NELEM = 33554432LL; // 32*64*128*128

// ---------- Tensor Data Mover (gfx1250) ----------
#if defined(__has_builtin)
#if __has_builtin(__builtin_amdgcn_tensor_load_to_lds) && __has_builtin(__builtin_amdgcn_s_wait_tensorcnt)
#define HAVE_TDM 1
#endif
#endif
#ifndef HAVE_TDM
#define HAVE_TDM 0
#endif

#if HAVE_TDM
typedef __attribute__((ext_vector_type(4))) unsigned int u32x4;
typedef __attribute__((ext_vector_type(8))) int          i32x8;
typedef __attribute__((ext_vector_type(4))) int          i32x4;

// 2-D TDM tile load: tile_d0 elements (4B) per row, tile_d1 rows, row stride
// stride0 elements. D# packing per CDNA5 ISA 8.3/8.4 (groups 2/3 zero: <=2D).
__device__ __forceinline__ void tdm_load_2d(unsigned lds_addr, unsigned long long gaddr,
                                            int tile_d0, int tile_d1, int stride0) {
  unsigned td0 = (unsigned)stride0;     // tensor_dim0 = row length
  unsigned td1 = (unsigned)tile_d1;     // tensor_dim1 = #rows
  u32x4 g0;
  g0[0] = 1u;                                                   // count=1, user D#
  g0[1] = lds_addr;                                             // lds_addr[31:0]
  g0[2] = (unsigned)(gaddr & 0xFFFFFFFFull);                    // global_addr[31:0]
  g0[3] = (unsigned)((gaddr >> 32) & 0x01FFFFFFull) | (2u << 30); // addr[56:32], type=2
  i32x8 g1;
  g1[0] = 0x00020000;                                           // data_size=4B, mask=0
  g1[1] = (int)((td0 & 0xFFFFu) << 16);                         // tensor_dim0[15:0]
  g1[2] = (int)((td0 >> 16) | ((td1 & 0xFFFFu) << 16));         // dim0 hi | tensor_dim1 lo
  g1[3] = (int)((td1 >> 16) | (((unsigned)tile_d0 & 0xFFFFu) << 16)); // dim1 hi | tile_dim0
  g1[4] = (int)((unsigned)tile_d1 & 0xFFFFu);                   // tile_dim1 (tile_dim2=0)
  g1[5] = stride0;                                              // tensor_dim0_stride[31:0]
  g1[6] = 0;
  g1[7] = 0;
  i32x4 z4 = {0, 0, 0, 0};
#if __clang_major__ >= 23
  i32x8 z8 = {0, 0, 0, 0, 0, 0, 0, 0};
  __builtin_amdgcn_tensor_load_to_lds(g0, g1, z4, z4, z8, 0);
#else
  __builtin_amdgcn_tensor_load_to_lds(g0, g1, z4, z4, 0);
#endif
}
#endif

// ---------- WMMA fragment gathers (wave32, contiguous k = 16*hi + e, same in A and B) ----------
// A (16-bit, 16x32) from row-major LDS tile t[row*ld + K].
__device__ __forceinline__ v16h frag_A(const _Float16* t, int ld) {
  int lane = threadIdx.x & 31;
  int row = lane & 15, hi = lane >> 4;
  v16h a;
#pragma unroll
  for (int e = 0; e < 16; ++e) {
    int K = e + hi * 16;
    a[e] = t[row * ld + K];
  }
  return a;
}
// A (16-bit, 16x32) from K-major f32 LDS tile: A[M][K] = t[K*ld + roff + M], scaled.
__device__ __forceinline__ v16h fragA_cm_f32(const float* t, int ld, int roff, float sgn) {
  int lane = threadIdx.x & 31;
  int row = lane & 15, hi = lane >> 4;
  v16h a;
#pragma unroll
  for (int e = 0; e < 16; ++e) {
    int K = e + hi * 16;
    a[e] = (_Float16)(sgn * t[K * ld + roff + row]);
  }
  return a;
}
// B (16-bit, 32x16) from K-major LDS tile t[K*ld + coff + col].
__device__ __forceinline__ v16h fragB_km(const _Float16* t, int ld, int coff) {
  int lane = threadIdx.x & 31;
  int col = lane & 15, hi = lane >> 4;
  v16h b;
#pragma unroll
  for (int e = 0; e < 16; ++e) {
    int K = e + hi * 16;
    b[e] = t[K * ld + coff + col];
  }
  return b;
}
// B (16-bit, 32x16) from N-major LDS tile: B[K][N] = t[(coff+N)*ld + K].
__device__ __forceinline__ v16h fragB_nm(const _Float16* t, int ld, int coff) {
  int lane = threadIdx.x & 31;
  int col = lane & 15, hi = lane >> 4;
  v16h b;
#pragma unroll
  for (int e = 0; e < 16; ++e) {
    int K = e + hi * 16;
    b[e] = t[(coff + col) * ld + K];
  }
  return b;
}

// ---------- ws small-region offsets (in floats) ----------
// [0..31] per-sample grad max, [32] q, [64..191] h, [192..319] r0.re, [320..447] r0.im,
// [448..575] g.re, [576..703] g.im, [1024..] Hr(16384), [17408..] Hi(16384),
// [33792..] Mr(128*129), [50304..] Mi(128*129). Big buffers start at 131072.

__global__ void k_init(float* W) {
  int t = threadIdx.x;
  if (t < 32) W[t] = 0.f;
}

// Depthwise Sobel |gx|+|gy|, per-sample max via block reduce + atomicMax (gm>=0).
__global__ void k_sobel(const float* __restrict__ x, float* W) {
  __shared__ float red[256];
  int tid = threadIdx.x;
  long long e = (long long)blockIdx.x * 256 + tid;
  int ww = (int)(e & 63); long long t = e >> 6;
  int hh = (int)(t & 127); t >>= 7;
  int ch = (int)(t & 127); int n = (int)(t >> 7);
  const float* base = x + ((long long)n * 128 + ch) * 128 * 64;
  float gx = 0.f, gy = 0.f;
#pragma unroll
  for (int dy = -1; dy <= 1; ++dy)
#pragma unroll
    for (int dx = -1; dx <= 1; ++dx) {
      int yh = hh + dy, xw = ww + dx;
      float v = 0.f;
      if (yh >= 0 && yh < 128 && xw >= 0 && xw < 64) v = base[yh * 64 + xw];
      float wx = (dx == 0) ? 0.f : ((dy == 0) ? 2.f : 1.f) * (float)dx;
      float wy = (dy == 0) ? 0.f : ((dx == 0) ? 2.f : 1.f) * (float)dy;
      gx += wx * v; gy += wy * v;
    }
  red[tid] = fabsf(gx) + fabsf(gy);
  __syncthreads();
  for (int s = 128; s > 0; s >>= 1) {
    if (tid < s) red[tid] = fmaxf(red[tid], red[tid + s]);
    __syncthreads();
  }
  if (tid == 0) atomicMax((unsigned int*)&W[n], __float_as_uint(red[0]));
}

// q = (1/(20*mean(max)/128^2))^0.25 * 5 ; h[j] = amp*exp(-pi*((j-64)/q)^2)
__global__ void k_qh(float* W) {
  __shared__ float qsh;
  int tid = threadIdx.x;
  if (tid == 0) {
    float s = 0.f;
    for (int i = 0; i < 32; ++i) s += W[i];
    float maxg = s / 32.f;
    float m = maxg / (128.f * 128.f);
    float q = powf(1.f / (20.f * m), 0.25f) * 5.f;
    W[32] = q; qsh = q;
  }
  __syncthreads();
  float q = qsh;
  float amp = sqrtf(sqrtf(2.f) / q);
  float d = ((float)tid - 64.f) / q;
  W[64 + tid] = amp * expf(-PI_F * d * d);
}

// H[m*16+nn][k] = amp*exp(-pi*(((k+16m)%128-64)/q)^2) * e^{i*2*pi*nn*k/16}
__global__ void k_buildH(float* W) {
  int gid = blockIdx.x * 256 + threadIdx.x; // 16384
  int row = gid >> 7, k = gid & 127;
  int m = row >> 4, nn = row & 15;
  float q = W[32];
  float amp = sqrtf(sqrtf(2.f) / q);
  int k1 = (k + m * 16) & 127;
  float dd = ((float)k1 - 64.f) / q;
  float ex = amp * expf(-PI_F * dd * dd);
  float ang = 2.f * PI_F * (float)((nn * k) & 15) / 16.f;
  W[1024 + gid]  = ex * cosf(ang);
  W[17408 + gid] = ex * sinf(ang);
}

// M = H @ conj(H), augmented with RHS e0 (V scale = L/(M*N) = 1)
__global__ void k_hhc(float* W) {
  const float* Hr = W + 1024; const float* Hi = W + 17408;
  int gid = blockIdx.x * 256 + threadIdx.x;
  int r = gid >> 7, c = gid & 127;
  float sr = 0.f, si = 0.f;
  for (int k = 0; k < 128; ++k) {
    float ar = Hr[r * 128 + k], ai = Hi[r * 128 + k];
    float br = Hr[k * 128 + c], bi = Hi[k * 128 + c];
    sr += ar * br + ai * bi;
    si += ai * br - ar * bi;
  }
  W[33792 + r * 129 + c] = sr;
  W[50304 + r * 129 + c] = si;
  if (c == 0) {
    W[33792 + r * 129 + 128] = (r == 0) ? 1.f : 0.f;
    W[50304 + r * 129 + 128] = 0.f;
  }
}

// One-workgroup complex Gaussian elimination (pinv@V == solve since V = e0),
// then r0 = conj(H) @ g.
__global__ void k_solve(float* W) {
  float* Mr = W + 33792; float* Mi = W + 50304;
  const float* Hr = W + 1024; const float* Hi = W + 17408;
  float* gr = W + 448; float* gi = W + 576;
  int t = threadIdx.x;
  for (int p = 0; p < 128; ++p) {
    __syncthreads();
    float pr = Mr[p * 129 + p], pi = Mi[p * 129 + p];
    float inv = 1.f / (pr * pr + pi * pi + 1e-30f);
    if (t > p) {
      float ar = Mr[t * 129 + p], ai = Mi[t * 129 + p];
      float fr = (ar * pr + ai * pi) * inv;
      float fi = (ai * pr - ar * pi) * inv;
      for (int qq = p; qq <= 128; ++qq) {
        float br = Mr[p * 129 + qq], bi = Mi[p * 129 + qq];
        Mr[t * 129 + qq] -= fr * br - fi * bi;
        Mi[t * 129 + qq] -= fr * bi + fi * br;
      }
    }
  }
  __syncthreads();
  if (t == 0) {
    for (int p = 127; p >= 0; --p) {
      float sr = Mr[p * 129 + 128], si = Mi[p * 129 + 128];
      for (int qq = p + 1; qq < 128; ++qq) {
        sr -= Mr[p * 129 + qq] * gr[qq] - Mi[p * 129 + qq] * gi[qq];
        si -= Mr[p * 129 + qq] * gi[qq] + Mi[p * 129 + qq] * gr[qq];
      }
      float pr = Mr[p * 129 + p], pi = Mi[p * 129 + p];
      float inv = 1.f / (pr * pr + pi * pi + 1e-30f);
      gr[p] = (sr * pr + si * pi) * inv;
      gi[p] = (si * pr - sr * pi) * inv;
    }
  }
  __syncthreads();
  float rr = 0.f, ri = 0.f;
  for (int k = 0; k < 128; ++k) {
    float ar = Hr[t * 128 + k], ai = Hi[t * 128 + k];
    rr += ar * gr[k] + ai * gi[k];
    ri += ar * gi[k] - ai * gr[k];
  }
  W[192 + t] = rr; W[320 + t] = ri;
}

// Stage 1: per (k,l) slice, R1[(b,c),(m,n)] = sum_{(i,j)} ag * S, S = r0[d(m,i)*32+k]*r0[d(n,j)*32+l]
// GEMM M=2048 N=16 K=16 (padded to 32) via v_wmma_f32_16x16x32_f16, real+imag.
__global__ void k_stage1(const float* __restrict__ x, const float* __restrict__ W,
                         float* __restrict__ R1r, float* __restrict__ R1i) {
  __shared__ _Float16 sBr[32 * 16];
  __shared__ _Float16 sBi[32 * 16];
  __shared__ _Float16 sA[8][16 * 32];
  const float* r0r = W + 192; const float* r0i = W + 320;
  int kk = blockIdx.x >> 5, ll = blockIdx.x & 31;
  int tid = threadIdx.x, lane = tid & 31, wave = tid >> 5;
  {
    int Krow = tid >> 4, Ncol = tid & 15;   // (i,j) x (m,n)
    int i = Krow >> 2, j = Krow & 3;
    int m = Ncol >> 2, n = Ncol & 3;
    int pa = ((i - m) & 3) * 32 + kk;
    int pb = ((j - n) & 3) * 32 + ll;
    float ar = r0r[pa], ai = r0i[pa];
    float br = r0r[pb], bi = r0i[pb];
    sBr[Krow * 16 + Ncol] = (_Float16)(ar * br - ai * bi);
    sBi[Krow * 16 + Ncol] = (_Float16)(ar * bi + ai * br);
    sBr[(Krow + 16) * 16 + Ncol] = (_Float16)0.f;   // K pad 16..31
    sBi[(Krow + 16) * 16 + Ncol] = (_Float16)0.f;
  }
  int Mtile = blockIdx.y * 8 + wave;     // 128 M-tiles over (b,c)
  int b = Mtile >> 2;
  int c0 = (Mtile & 3) * 16;
#pragma unroll
  for (int s = 0; s < 16; ++s) {
    int pos = s * 32 + lane;
    int row = pos >> 5, K = pos & 31;
    float v = 0.f;
    if (K < 16) {
      int i = K >> 2, j = K & 3;
      long long addr = ((((long long)(b * 4 + i) * 32 + kk) * 4 + j) * 32 + ll) * 64 + (c0 + row);
      v = x[addr];
    }
    sA[wave][row * 32 + K] = (_Float16)v;
  }
  __syncthreads();
  v16h a  = frag_A(&sA[wave][0], 32);
  v16h br = fragB_km(sBr, 16, 0);
  v16h bi = fragB_km(sBi, 16, 0);
  v8f cr = {0, 0, 0, 0, 0, 0, 0, 0}, ci = {0, 0, 0, 0, 0, 0, 0, 0};
  cr = __builtin_amdgcn_wmma_f32_16x16x32_f16(false, a, false, br, (short)0, cr, false, false);
  ci = __builtin_amdgcn_wmma_f32_16x16x32_f16(false, a, false, bi, (short)0, ci, false, false);
  int N = lane & 15, hi = lane >> 4;
  int m = N >> 2, n = N & 3;
  long long obase = ((((long long)(b * 4 + m) * 4 + n) * 32 + kk) * 32 + ll) * 64 + c0 + hi * 8;
#pragma unroll
  for (int r = 0; r < 8; ++r) { R1r[obase + r] = cr[r]; R1i[obase + r] = ci[r]; }
}

// fft2 + (Re - Im) folded: pass 1 computes E = Rr*P - Ri*Q, F = Rr*Q + Ri*P per
// (b,m,n,k) block. Batched 32x32x32 GEMMs on WMMA, K=32 exact. Input tile is a
// contiguous 8KB block -> staged by the Tensor Data Mover (one DMA per buffer),
// f32->f16 conversion folded into the fragment gather.
__global__ void k_cas_rows(const float* __restrict__ Sr, const float* __restrict__ Si,
                           float* __restrict__ E, float* __restrict__ F, int complexIn) {
  __shared__ float sRf[2048];
  __shared__ float sIf[2048];
  __shared__ _Float16 sP[32 * 32];
  __shared__ _Float16 sQ[32 * 32];
  int tid = threadIdx.x, lane = tid & 31, wave = tid >> 5;
  int kk = blockIdx.x & 31;
  long long bmn = blockIdx.x >> 5;
  long long base = (bmn * 32 + kk) * 2048;
#pragma unroll
  for (int s = 0; s < 4; ++s) {
    int e = tid + s * 256;
    int l = e >> 5, qv = e & 31;
    float th = 2.f * PI_F * (float)((l * qv) & 31) / 32.f;
    float cv = cosf(th), sv = sinf(th);
    sP[e] = (_Float16)(cv + sv); sQ[e] = (_Float16)(cv - sv);
  }
  __builtin_prefetch(&Sr[base + 2048]);            // next k-slice -> L2 (speculative)
#if HAVE_TDM
  if (tid < 32) {                                  // one wave issues the DMA (EXEC ignored)
    tdm_load_2d((unsigned)(unsigned long long)(void*)sRf,
                (unsigned long long)(const void*)(Sr + base), 2048, 1, 2048);
    if (complexIn)
      tdm_load_2d((unsigned)(unsigned long long)(void*)sIf,
                  (unsigned long long)(const void*)(Si + base), 2048, 1, 2048);
    __builtin_amdgcn_s_wait_tensorcnt(0);
  }
#else
#pragma unroll
  for (int s = 0; s < 8; ++s) {
    int e = tid + s * 256;
    sRf[e] = Sr[base + e];
    if (complexIn) sIf[e] = Si[base + e];
  }
#endif
  __syncthreads();
  int c0 = (wave & 3) * 16;      // 4 c-tiles
  int q0 = (wave >> 2) * 16;     // 2 q-tiles
  v16h Ar = fragA_cm_f32(sRf, 64, c0, 1.f);        // A[c][l]
  v16h Bp = fragB_km(sP, 32, q0);                  // B[l][q]
  v16h Bq = fragB_km(sQ, 32, q0);
  v8f accE = {0, 0, 0, 0, 0, 0, 0, 0}, accF = {0, 0, 0, 0, 0, 0, 0, 0};
  accE = __builtin_amdgcn_wmma_f32_16x16x32_f16(false, Ar, false, Bp, (short)0, accE, false, false);
  accF = __builtin_amdgcn_wmma_f32_16x16x32_f16(false, Ar, false, Bq, (short)0, accF, false, false);
  if (complexIn) {
    v16h Ain = fragA_cm_f32(sIf, 64, c0, -1.f);
    v16h Aip = fragA_cm_f32(sIf, 64, c0, 1.f);
    accE = __builtin_amdgcn_wmma_f32_16x16x32_f16(false, Ain, false, Bq, (short)0, accE, false, false);
    accF = __builtin_amdgcn_wmma_f32_16x16x32_f16(false, Aip, false, Bp, (short)0, accF, false, false);
  }
  int N = lane & 15, hi = lane >> 4;               // D: M=c offset, N=q offset
  long long o = base + (long long)(q0 + N) * 64 + c0 + hi * 8;
#pragma unroll
  for (int r = 0; r < 8; ++r) { E[o + r] = accE[r]; F[o + r] = accF[r]; }
}

// Pass 2: out[p][c] = sum_k C[p,k]*E[k][c] + S[p,k]*F[k][c] per (b,m,n,q) block.
// Input tiles are strided 2-D (32 rows x 256B, stride 8KB) -> TDM 2-D tile load.
__global__ void k_cas_cols(const float* __restrict__ E, const float* __restrict__ F,
                           float* __restrict__ out) {
  __shared__ float sEf[2048];
  __shared__ float sFf[2048];
  __shared__ _Float16 sC[32 * 32];
  __shared__ _Float16 sS[32 * 32];
  int tid = threadIdx.x, lane = tid & 31, wave = tid >> 5;
  int qv = blockIdx.x & 31;
  long long bmn = blockIdx.x >> 5;
  long long blk = bmn * 65536;
#pragma unroll
  for (int s = 0; s < 4; ++s) {
    int e = tid + s * 256;
    int p = e >> 5, k = e & 31;
    float th = 2.f * PI_F * (float)((p * k) & 31) / 32.f;
    sC[e] = (_Float16)cosf(th); sS[e] = (_Float16)sinf(th);
  }
#if HAVE_TDM
  if (tid < 32) {
    tdm_load_2d((unsigned)(unsigned long long)(void*)sEf,
                (unsigned long long)(const void*)(E + blk + (long long)qv * 64), 64, 32, 2048);
    tdm_load_2d((unsigned)(unsigned long long)(void*)sFf,
                (unsigned long long)(const void*)(F + blk + (long long)qv * 64), 64, 32, 2048);
    __builtin_amdgcn_s_wait_tensorcnt(0);
  }
#else
#pragma unroll
  for (int s = 0; s < 8; ++s) {
    int e = tid + s * 256;
    int k = e >> 6, cc = e & 63;
    long long src = blk + ((long long)k * 32 + qv) * 64 + cc;
    sEf[e] = E[src]; sFf[e] = F[src];
  }
#endif
  __syncthreads();
  int c0 = (wave & 3) * 16;
  int p0 = (wave >> 2) * 16;
  v16h Ae = fragA_cm_f32(sEf, 64, c0, 1.f);        // A[c][k]
  v16h Af = fragA_cm_f32(sFf, 64, c0, 1.f);
  v16h Bc = fragB_nm(sC, 32, p0);                  // B[k][p] = sC[p*32+k]
  v16h Bs = fragB_nm(sS, 32, p0);
  v8f acc = {0, 0, 0, 0, 0, 0, 0, 0};
  acc = __builtin_amdgcn_wmma_f32_16x16x32_f16(false, Ae, false, Bc, (short)0, acc, false, false);
  acc = __builtin_amdgcn_wmma_f32_16x16x32_f16(false, Af, false, Bs, (short)0, acc, false, false);
  int N = lane & 15, hi = lane >> 4;               // M=c offset, N=p offset
  int cc = c0 + hi * 8;
  long long o = blk + ((long long)(p0 + N) * 32 + qv) * 64 + cc;
#pragma unroll
  for (int r = 0; r < 8; ++r) out[o + r] = acc[r];
}

// Block-diagonal MLP: 4 groups of 16 features (two 8-blocks each); two WMMA
// layers with fused bias+ReLU and bias+softshrink, LDS round-trip between layers.
__global__ void k_mlp(const float* __restrict__ xb,
                      const float* __restrict__ w1, const float* __restrict__ b1,
                      const float* __restrict__ w2, const float* __restrict__ b2,
                      float* __restrict__ xs) {
  __shared__ _Float16 sW1[4][32 * 16];
  __shared__ _Float16 sW2[4][32 * 16];
  __shared__ _Float16 sA[8][16 * 32];
  int tid = threadIdx.x, lane = tid & 31, wave = tid >> 5;
#pragma unroll
  for (int s = 0; s < 8; ++s) {
    int e = tid + s * 256;            // 2048 entries
    int g = e >> 9;
    int K = (e >> 4) & 31;
    int Ncol = e & 15;
    _Float16 v1 = (_Float16)0.f, v2 = (_Float16)0.f;
    if (K < 16 && (K >> 3) == (Ncol >> 3)) {
      int blk = g * 2 + (K >> 3);
      int ii = K & 7, oo = Ncol & 7;
      v1 = (_Float16)w1[(blk * 8 + ii) * 8 + oo];
      v2 = (_Float16)w2[(blk * 8 + ii) * 8 + oo];
    }
    sW1[g][K * 16 + Ncol] = v1;
    sW2[g][K * 16 + Ncol] = v2;
  }
  long long Mtile = (long long)blockIdx.x * 8 + wave;
  long long row0 = Mtile * 16;
  int N = lane & 15, hi = lane >> 4;
  for (int g = 0; g < 4; ++g) {
    __syncthreads();
#pragma unroll
    for (int s = 0; s < 16; ++s) {
      int pos = s * 32 + lane;
      int row = pos >> 5, K = pos & 31;
      float v = (K < 16) ? xb[(row0 + row) * 64 + g * 16 + K] : 0.f;
      sA[wave][row * 32 + K] = (_Float16)v;
    }
    __syncthreads();
    v16h a  = frag_A(&sA[wave][0], 32);
    v16h bw = fragB_km(&sW1[g][0], 16, 0);
    v8f acc = {0, 0, 0, 0, 0, 0, 0, 0};
    acc = __builtin_amdgcn_wmma_f32_16x16x32_f16(false, a, false, bw, (short)0, acc, false, false);
    int blk = g * 2 + (N >> 3), oo = N & 7;
    float bb1 = b1[blk * 8 + oo], bb2 = b2[blk * 8 + oo];
    __syncthreads();
#pragma unroll
    for (int r = 0; r < 8; ++r) {
      float o1 = fmaxf(acc[r] + bb1, 0.f);
      int M = r + hi * 8;
      sA[wave][M * 32 + N] = (_Float16)o1;   // K cols 16..31 stay zero
    }
    __syncthreads();
    v16h a2  = frag_A(&sA[wave][0], 32);
    v16h bw2 = fragB_km(&sW2[g][0], 16, 0);
    v8f acc2 = {0, 0, 0, 0, 0, 0, 0, 0};
    acc2 = __builtin_amdgcn_wmma_f32_16x16x32_f16(false, a2, false, bw2, (short)0, acc2, false, false);
#pragma unroll
    for (int r = 0; r < 8; ++r) {
      int M = r + hi * 8;
      float o2 = acc2[r] + bb2;
      float sv = (o2 > 0.01f) ? (o2 - 0.01f) : ((o2 < -0.01f) ? (o2 + 0.01f) : 0.f);
      xs[(row0 + M) * 64 + g * 16 + N] = sv;
    }
    __syncthreads();
  }
}

// Final: per (k,l): I[(b,c),(x,y)] = sum_{(m,n)} aa * h[d(m,x)*32+l]*h[d(n,y)*32+l],
// fused residual add with original x.
__global__ void k_final(const float* __restrict__ aa, const float* __restrict__ W,
                        const float* __restrict__ xin, float* __restrict__ out) {
  __shared__ _Float16 sB[32 * 16];
  __shared__ _Float16 sA[8][16 * 32];
  const float* h = W + 64;
  int kk = blockIdx.x >> 5, ll = blockIdx.x & 31;
  int tid = threadIdx.x, lane = tid & 31, wave = tid >> 5;
  {
    int Krow = tid >> 4, Ncol = tid & 15;  // (m,n) x (x,y)
    int m = Krow >> 2, n = Krow & 3;
    int xx = Ncol >> 2, yy = Ncol & 3;
    float v = h[((xx - m) & 3) * 32 + ll] * h[((yy - n) & 3) * 32 + ll];
    sB[Krow * 16 + Ncol] = (_Float16)v;
    sB[(Krow + 16) * 16 + Ncol] = (_Float16)0.f;
  }
  int Mtile = blockIdx.y * 8 + wave;
  int b = Mtile >> 2;
  int c0 = (Mtile & 3) * 16;
#pragma unroll
  for (int s = 0; s < 16; ++s) {
    int pos = s * 32 + lane;
    int row = pos >> 5, K = pos & 31;
    float v = 0.f;
    if (K < 16) {
      int m = K >> 2, n = K & 3;
      long long addr = ((((long long)(b * 4 + m) * 4 + n) * 32 + kk) * 32 + ll) * 64 + (c0 + row);
      v = aa[addr];
    }
    sA[wave][row * 32 + K] = (_Float16)v;
  }
  __syncthreads();
  v16h a  = frag_A(&sA[wave][0], 32);
  v16h bf = fragB_km(sB, 16, 0);
  v8f acc = {0, 0, 0, 0, 0, 0, 0, 0};
  acc = __builtin_amdgcn_wmma_f32_16x16x32_f16(false, a, false, bf, (short)0, acc, false, false);
  int N = lane & 15, hi = lane >> 4;
  int xx = N >> 2, yy = N & 3;
  long long obase = ((((long long)(b * 4 + xx) * 32 + kk) * 4 + yy) * 32 + ll) * 64 + c0 + hi * 8;
#pragma unroll
  for (int r = 0; r < 8; ++r) out[obase + r] = acc[r] + xin[obase + r];
}

extern "C" void kernel_launch(void* const* d_in, const int* in_sizes, int n_in,
                              void* d_out, int out_size, void* d_ws, size_t ws_size,
                              hipStream_t stream) {
  (void)in_sizes; (void)n_in; (void)out_size; (void)ws_size;
  const float* x  = (const float*)d_in[0];
  const float* w1 = (const float*)d_in[1];
  const float* b1 = (const float*)d_in[2];
  const float* w2 = (const float*)d_in[3];
  const float* b2 = (const float*)d_in[4];
  float* out = (float*)d_out;
  float* W = (float*)d_ws;
  float* bufA = W + 131072;
  float* bufB = bufA + NELEM;
  float* bufC = bufB + NELEM;
  float* bufD = bufC + NELEM;

  k_init  <<<1,      64,  0, stream>>>(W);
  k_sobel <<<131072, 256, 0, stream>>>(x, W);
  k_qh    <<<1,      128, 0, stream>>>(W);
  k_buildH<<<64,     256, 0, stream>>>(W);
  k_hhc   <<<64,     256, 0, stream>>>(W);
  k_solve <<<1,      128, 0, stream>>>(W);
  // R1 (complex) into bufA/bufB
  k_stage1<<<dim3(1024, 16), 256, 0, stream>>>(x, W, bufA, bufB);
  // fft2 + (Re-Im): two WMMA passes -> xb into bufA
  k_cas_rows<<<16384, 256, 0, stream>>>(bufA, bufB, bufC, bufD, 1);
  k_cas_cols<<<16384, 256, 0, stream>>>(bufC, bufD, bufA);
  // block-diag MLP + softshrink -> xs into bufB
  k_mlp<<<4096, 256, 0, stream>>>(bufA, w1, b1, w2, b2, bufB);
  // fft2 + (Re-Im) on real xs -> aa into bufA
  k_cas_rows<<<16384, 256, 0, stream>>>(bufB, bufB, bufC, bufD, 0);
  k_cas_cols<<<16384, 256, 0, stream>>>(bufC, bufD, bufA);
  // synthesis + residual
  k_final<<<dim3(1024, 16), 256, 0, stream>>>(bufA, W, x, out);
}
